// MoEFFN_18176301597567
// MI455X (gfx1250) — compile-verified, matrix operand order checked by reading, not run on hardware
//
#include <hip/hip_runtime.h>
#include <hip/hip_bf16.h>
#include <stdint.h>

typedef __bf16 bf16;
typedef __attribute__((ext_vector_type(16))) __bf16 v16bf;
typedef __attribute__((ext_vector_type(8)))  float  v8f;

namespace {
constexpr int S_TOK  = 4096;   // B*T
constexpr int C_DIM  = 512;
constexpr int E_NUM  = 64;
constexpr int G_NUM  = 8;
constexpr int EPG    = E_NUM / G_NUM;   // 8
constexpr int TG_SEL = 4;
constexpr int K_TOP  = 6;
constexpr int H_DIM  = 160;
constexpr int HS_DIM = 512;
constexpr int NSLOT  = S_TOK * K_TOP;   // 24576
}

// ---------------- WMMA helpers (CDNA5 wave32, 16x16x32 bf16, f32 acc) ------

__device__ __forceinline__ v8f wmma_bf16(v16bf a, v16bf b, v8f c) {
  return __builtin_amdgcn_wmma_f32_16x16x32_bf16(
      /*neg_a=*/false, a, /*neg_b=*/false, b,
      /*c_mod=*/(short)0, c, /*reuse_a=*/false, /*reuse_b=*/false);
}

// 16-bit A/B fragment, row-major tile origin p, leading dim `stride`.
// ISA layout: lanes 0-15 row=lane, Kbase=0; lanes 16-31 row=lane-16, Kbase=8.
// vector elems (2j,2j+1): K = Kbase + 2j (j<4), K = 16 + Kbase + 2(j-4) (j>=4).
__device__ __forceinline__ v16bf load_frag(const bf16* __restrict__ p, int stride) {
  const int lane = threadIdx.x & 31;
  const int row  = lane & 15;
  const int kb   = (lane < 16) ? 0 : 8;
  const bf16* q = p + (size_t)row * stride + kb;
  v16bf f;
#pragma unroll
  for (int j = 0; j < 4; ++j) { f[2*j]   = q[2*j];      f[2*j+1] = q[2*j+1]; }
#pragma unroll
  for (int j = 0; j < 4; ++j) { f[8+2*j] = q[16+2*j];   f[9+2*j] = q[17+2*j]; }
  return f;
}

__device__ __forceinline__ v16bf load_frag_clamp(const bf16* __restrict__ p, int stride,
                                                 int nrows) {
  const int lane = threadIdx.x & 31;
  int row = lane & 15;
  row = (row < nrows) ? row : (nrows - 1);
  const int kb = (lane < 16) ? 0 : 8;
  const bf16* q = p + (size_t)row * stride + kb;
  v16bf f;
#pragma unroll
  for (int j = 0; j < 4; ++j) { f[2*j]   = q[2*j];      f[2*j+1] = q[2*j+1]; }
#pragma unroll
  for (int j = 0; j < 4; ++j) { f[8+2*j] = q[16+2*j];   f[9+2*j] = q[17+2*j]; }
  return f;
}

// Gathered A fragment: rows indexed through sorted token list.
__device__ __forceinline__ v16bf load_frag_gather(const bf16* __restrict__ xb,
                                                  const int* __restrict__ rows,
                                                  int nrows, int k0) {
  const int lane = threadIdx.x & 31;
  const int r    = lane & 15;
  const int kb   = (lane < 16) ? 0 : 8;
  const int tok  = rows[(r < nrows) ? r : 0];
  const bf16* q = xb + (size_t)tok * C_DIM + k0 + kb;
  v16bf f;
#pragma unroll
  for (int j = 0; j < 4; ++j) { f[2*j]   = q[2*j];      f[2*j+1] = q[2*j+1]; }
#pragma unroll
  for (int j = 0; j < 4; ++j) { f[8+2*j] = q[16+2*j];   f[9+2*j] = q[17+2*j]; }
  return f;
}

// fast sigmoid / silu: v_exp_f32 + v_rcp_f32 (no IEEE divide expansion)
__device__ __forceinline__ float sigmoid_f(float x) {
  return __builtin_amdgcn_rcpf(1.0f + __expf(-x));
}
__device__ __forceinline__ float silu_f(float x) {
  return x * sigmoid_f(x);
}

// ---------------- conversion kernels ---------------------------------------

__global__ void cvt_x_kernel(const float* __restrict__ x, bf16* __restrict__ xb, int n) {
  int i = blockIdx.x * blockDim.x + threadIdx.x;
  if (i < n) xb[i] = (bf16)x[i];
}

// W: [nE][R][Cc] fp32  ->  WT: [nE][Cc][R] bf16   (N-major for B fragments)
__global__ void cvt_transpose_kernel(const float* __restrict__ W, bf16* __restrict__ WT,
                                     int nE, int R, int Cc) {
  size_t i = (size_t)blockIdx.x * blockDim.x + threadIdx.x;
  size_t per = (size_t)R * Cc;
  size_t total = (size_t)nE * per;
  if (i >= total) return;
  int e = (int)(i / per);
  size_t rem = i - (size_t)e * per;
  int r = (int)(rem / Cc);
  int c = (int)(rem - (size_t)r * Cc);
  WT[(size_t)e * per + (size_t)c * R + r] = (bf16)W[i];
}

// ---------------- router ----------------------------------------------------

__global__ void router_kernel(const float* __restrict__ x,
                              const float* __restrict__ rw,
                              const float* __restrict__ bias,
                              int* __restrict__ topk_id,
                              float* __restrict__ topk_w,
                              int* __restrict__ counts) {
  const int s = blockIdx.x;
  const int e = threadIdx.x;           // 64 threads = 64 experts
  __shared__ float sc[E_NUM], bi[E_NUM], gm[G_NUM];

  const float* xr = x + (size_t)s * C_DIM;
  const float* wr = rw + (size_t)e * C_DIM;
  float acc = 0.f;
#pragma unroll 4
  for (int c = 0; c < C_DIM; ++c) acc = fmaf(xr[c], wr[c], acc);
  float sco = sigmoid_f(acc);
  sc[e] = sco;
  bi[e] = sco + bias[e];
  __syncthreads();

  if (e < G_NUM) {
    float m = bi[e * EPG];
#pragma unroll
    for (int j = 1; j < EPG; ++j) m = fmaxf(m, bi[e * EPG + j]);
    gm[e] = m;
  }
  __syncthreads();

  if (e == 0) {
    float g[G_NUM];
    bool  gsel[G_NUM];
#pragma unroll
    for (int i = 0; i < G_NUM; ++i) { g[i] = gm[i]; gsel[i] = false; }
    for (int t = 0; t < TG_SEL; ++t) {           // top-TG groups, first-wins ties
      int bidx = 0; float bv = -1e30f;
      for (int i = 0; i < G_NUM; ++i) if (g[i] > bv) { bv = g[i]; bidx = i; }
      gsel[bidx] = true; g[bidx] = -1e30f;
    }
    float mk[E_NUM];
    for (int i = 0; i < E_NUM; ++i) mk[i] = gsel[i / EPG] ? bi[i] : -1e30f;
    int ids[K_TOP];
    float wsum = 0.f;
    for (int t = 0; t < K_TOP; ++t) {            // top-K experts inside groups
      int bidx = 0; float bv = -1e30f;
      for (int i = 0; i < E_NUM; ++i) if (mk[i] > bv) { bv = mk[i]; bidx = i; }
      ids[t] = bidx; mk[bidx] = -1e30f;
      wsum += sc[bidx];                          // gate from UNbiased scores
    }
    float inv = __builtin_amdgcn_rcpf(wsum + 1e-20f);
    for (int t = 0; t < K_TOP; ++t) {
      topk_id[s * K_TOP + t] = ids[t];
      topk_w[s * K_TOP + t]  = sc[ids[t]] * inv;
      atomicAdd(&counts[ids[t]], 1);
    }
  }
}

__global__ void scan_kernel(const int* __restrict__ counts,
                            int* __restrict__ offsets,
                            int* __restrict__ cursors) {
  if (threadIdx.x == 0) {
    int acc = 0;
    for (int e = 0; e < E_NUM; ++e) { offsets[e] = acc; cursors[e] = acc; acc += counts[e]; }
    offsets[E_NUM] = acc;
  }
}

__global__ void scatter_kernel(const int* __restrict__ topk_id,
                               const float* __restrict__ topk_w,
                               int* __restrict__ cursors,
                               int* __restrict__ sorted_tok,
                               float* __restrict__ slot_gate,
                               int* __restrict__ inv_slot) {
  int s = blockIdx.x * blockDim.x + threadIdx.x;
  if (s >= S_TOK) return;
  for (int k = 0; k < K_TOP; ++k) {
    int e = topk_id[s * K_TOP + k];
    int slot = atomicAdd(&cursors[e], 1);
    sorted_tok[slot] = s;
    slot_gate[slot]  = topk_w[s * K_TOP + k];
    inv_slot[s * K_TOP + k] = slot;
  }
}

// ---------------- shared expert GEMMs ---------------------------------------
// Each wave computes a 16(M)x32(N) output slab: A fragment reused by 2 N tiles.
// shared_up: grid(S/16, HS/256), block 256 (8 waves).

__global__ void shared_up_kernel(const bf16* __restrict__ xb,
                                 const bf16* __restrict__ gT,   // [HS][C]
                                 const bf16* __restrict__ uT,   // [HS][C]
                                 bf16* __restrict__ hs) {       // [S][HS]
  const int wave = threadIdx.x >> 5;
  const int lane = threadIdx.x & 31;
  const int m0 = blockIdx.x * 16;
  const int n0 = (blockIdx.y * 8 + wave) * 32;
  const bf16* ap  = xb + (size_t)m0 * C_DIM;
  const bf16* bg0 = gT + (size_t)n0 * C_DIM;
  const bf16* bg1 = bg0 + (size_t)16 * C_DIM;
  const bf16* bu0 = uT + (size_t)n0 * C_DIM;
  const bf16* bu1 = bu0 + (size_t)16 * C_DIM;
  v8f cg0 = {}, cg1 = {}, cu0 = {}, cu1 = {};
  for (int k0 = 0; k0 < C_DIM; k0 += 32) {
    __builtin_prefetch(bg0 + k0 + 128, 0, 3);
    __builtin_prefetch(bu0 + k0 + 128, 0, 3);
    v16bf a  = load_frag(ap + k0, C_DIM);
    v16bf b0 = load_frag(bg0 + k0, C_DIM);
    v16bf b1 = load_frag(bg1 + k0, C_DIM);
    v16bf b2 = load_frag(bu0 + k0, C_DIM);
    v16bf b3 = load_frag(bu1 + k0, C_DIM);
    cg0 = wmma_bf16(a, b0, cg0);
    cg1 = wmma_bf16(a, b1, cg1);
    cu0 = wmma_bf16(a, b2, cu0);
    cu1 = wmma_bf16(a, b3, cu1);
  }
  const int n  = lane & 15;
  const int rb = (lane < 16) ? 0 : 8;
#pragma unroll
  for (int r = 0; r < 8; ++r) {
    size_t rowo = (size_t)(m0 + rb + r) * HS_DIM + n0 + n;
    hs[rowo]      = (bf16)(silu_f(cg0[r]) * cu0[r]);
    hs[rowo + 16] = (bf16)(silu_f(cg1[r]) * cu1[r]);
  }
}

// shared_down: grid(S/16, C/256), block 256.
__global__ void shared_down_kernel(const bf16* __restrict__ hs,   // [S][HS]
                                   const bf16* __restrict__ dT,   // [C][HS]
                                   float* __restrict__ out) {     // [S][C]
  const int wave = threadIdx.x >> 5;
  const int lane = threadIdx.x & 31;
  const int m0 = blockIdx.x * 16;
  const int n0 = (blockIdx.y * 8 + wave) * 32;
  const bf16* ap  = hs + (size_t)m0 * HS_DIM;
  const bf16* bp0 = dT + (size_t)n0 * HS_DIM;
  const bf16* bp1 = bp0 + (size_t)16 * HS_DIM;
  v8f c0 = {}, c1 = {};
  for (int k0 = 0; k0 < HS_DIM; k0 += 32) {
    __builtin_prefetch(bp0 + k0 + 128, 0, 3);
    v16bf a  = load_frag(ap + k0, HS_DIM);
    v16bf b0 = load_frag(bp0 + k0, HS_DIM);
    v16bf b1 = load_frag(bp1 + k0, HS_DIM);
    c0 = wmma_bf16(a, b0, c0);
    c1 = wmma_bf16(a, b1, c1);
  }
  const int n  = lane & 15;
  const int rb = (lane < 16) ? 0 : 8;
#pragma unroll
  for (int r = 0; r < 8; ++r) {
    size_t rowo = (size_t)(m0 + rb + r) * C_DIM + n0 + n;
    out[rowo]      = c0[r];      // overwrite (fresh each call)
    out[rowo + 16] = c1[r];
  }
}

// ---------------- routed experts (grouped sparse dispatch) ------------------
// up: grid(E, H/32), block 256; each wave strides over the expert's M tiles,
// computing a 16x32 slab (A fragment shared by 4 WMMAs: g0,g1,u0,u1).

__global__ void routed_up_kernel(const bf16* __restrict__ xb,       // [S][C]
                                 const bf16* __restrict__ gT,       // [E][H][C]
                                 const bf16* __restrict__ uT,       // [E][H][C]
                                 const int* __restrict__ offsets,
                                 const int* __restrict__ sorted_tok,
                                 const float* __restrict__ slot_gate,
                                 bf16* __restrict__ hbuf) {         // [NSLOT][H]
  const int e    = blockIdx.x;
  const int n0   = blockIdx.y * 32;
  const int wave = threadIdx.x >> 5;
  const int lane = threadIdx.x & 31;
  const int off  = offsets[e];
  const int ne   = offsets[e + 1] - off;
  const bf16* bg0 = gT + (size_t)e * H_DIM * C_DIM + (size_t)n0 * C_DIM;
  const bf16* bg1 = bg0 + (size_t)16 * C_DIM;
  const bf16* bu0 = uT + (size_t)e * H_DIM * C_DIM + (size_t)n0 * C_DIM;
  const bf16* bu1 = bu0 + (size_t)16 * C_DIM;

  for (int m0 = wave * 16; m0 < ne; m0 += 16 * 8) {
    const int* rows = sorted_tok + off + m0;
    const int nr = ne - m0;
    v8f cg0 = {}, cg1 = {}, cu0 = {}, cu1 = {};
    for (int k0 = 0; k0 < C_DIM; k0 += 32) {
      __builtin_prefetch(bg0 + k0 + 128, 0, 3);
      v16bf a  = load_frag_gather(xb, rows, nr, k0);
      v16bf b0 = load_frag(bg0 + k0, C_DIM);
      v16bf b1 = load_frag(bg1 + k0, C_DIM);
      v16bf b2 = load_frag(bu0 + k0, C_DIM);
      v16bf b3 = load_frag(bu1 + k0, C_DIM);
      cg0 = wmma_bf16(a, b0, cg0);
      cg1 = wmma_bf16(a, b1, cg1);
      cu0 = wmma_bf16(a, b2, cu0);
      cu1 = wmma_bf16(a, b3, cu1);
    }
    const int n  = lane & 15;
    const int rb = (lane < 16) ? 0 : 8;
#pragma unroll
    for (int r = 0; r < 8; ++r) {
      int m = m0 + rb + r;
      if (m < ne) {
        int slot = off + m;
        float gate = slot_gate[slot];
        size_t rowo = (size_t)slot * H_DIM + n0 + n;
        hbuf[rowo]      = (bf16)(silu_f(cg0[r]) * cu0[r] * gate);
        hbuf[rowo + 16] = (bf16)(silu_f(cg1[r]) * cu1[r] * gate);
      }
    }
  }
}

// down: grid(E, C/32), block 256; 16x32 slab per wave.
__global__ void routed_down_kernel(const bf16* __restrict__ hbuf,   // [NSLOT][H]
                                   const bf16* __restrict__ dT,     // [E][C][H]
                                   const int* __restrict__ offsets,
                                   float* __restrict__ yslot) {     // [NSLOT][C]
  const int e    = blockIdx.x;
  const int n0   = blockIdx.y * 32;
  const int wave = threadIdx.x >> 5;
  const int lane = threadIdx.x & 31;
  const int off  = offsets[e];
  const int ne   = offsets[e + 1] - off;
  const bf16* bp0 = dT + (size_t)e * C_DIM * H_DIM + (size_t)n0 * H_DIM;
  const bf16* bp1 = bp0 + (size_t)16 * H_DIM;

  for (int m0 = wave * 16; m0 < ne; m0 += 16 * 8) {
    const bf16* ap = hbuf + (size_t)(off + m0) * H_DIM;
    const int nr = ne - m0;
    v8f c0 = {}, c1 = {};
    for (int k0 = 0; k0 < H_DIM; k0 += 32) {       // H=160 -> 5 iterations
      v16bf a  = load_frag_clamp(ap + k0, H_DIM, nr);
      v16bf b0 = load_frag(bp0 + k0, H_DIM);
      v16bf b1 = load_frag(bp1 + k0, H_DIM);
      c0 = wmma_bf16(a, b0, c0);
      c1 = wmma_bf16(a, b1, c1);
    }
    const int n  = lane & 15;
    const int rb = (lane < 16) ? 0 : 8;
#pragma unroll
    for (int r = 0; r < 8; ++r) {
      int m = m0 + rb + r;
      if (m < ne) {
        size_t rowo = (size_t)(off + m) * C_DIM + n0 + n;
        yslot[rowo]      = c0[r];
        yslot[rowo + 16] = c1[r];
      }
    }
  }
}

// Deterministic combine: out(s,:) = shared(s,:) + sum_k yslot[inv_slot[s,k],:]
__global__ void combine_kernel(const float* __restrict__ yslot,
                               const int* __restrict__ inv_slot,
                               float* __restrict__ out) {
  const int s = blockIdx.x;
  int slots[K_TOP];
#pragma unroll
  for (int k = 0; k < K_TOP; ++k) slots[k] = inv_slot[s * K_TOP + k];
  for (int c = threadIdx.x; c < C_DIM; c += blockDim.x) {
    float acc = out[(size_t)s * C_DIM + c];
#pragma unroll
    for (int k = 0; k < K_TOP; ++k)
      acc += yslot[(size_t)slots[k] * C_DIM + c];
    out[(size_t)s * C_DIM + c] = acc;
  }
}

// ---------------- host launch -----------------------------------------------

extern "C" void kernel_launch(void* const* d_in, const int* in_sizes, int n_in,
                              void* d_out, int out_size, void* d_ws, size_t ws_size,
                              hipStream_t stream) {
  const float* x    = (const float*)d_in[0];
  const float* rw   = (const float*)d_in[1];
  const float* bias = (const float*)d_in[2];
  const float* Wg   = (const float*)d_in[3];   // [E][C][H]
  const float* Wu   = (const float*)d_in[4];   // [E][C][H]
  const float* Wd   = (const float*)d_in[5];   // [E][H][C]
  const float* sWg  = (const float*)d_in[6];   // [C][HS]
  const float* sWu  = (const float*)d_in[7];   // [C][HS]
  const float* sWd  = (const float*)d_in[8];   // [HS][C]
  float* out = (float*)d_out;

  size_t o = 0;
  auto alloc = [&](size_t bytes) -> void* {
    void* p = (char*)d_ws + o;
    o += (bytes + 255) & ~(size_t)255;
    return p;
  };
  bf16* xb     = (bf16*)alloc((size_t)S_TOK * C_DIM * 2);
  bf16* WgT    = (bf16*)alloc((size_t)E_NUM * H_DIM * C_DIM * 2);
  bf16* WuT    = (bf16*)alloc((size_t)E_NUM * H_DIM * C_DIM * 2);
  bf16* WdT    = (bf16*)alloc((size_t)E_NUM * C_DIM * H_DIM * 2);
  bf16* sWgT   = (bf16*)alloc((size_t)HS_DIM * C_DIM * 2);
  bf16* sWuT   = (bf16*)alloc((size_t)HS_DIM * C_DIM * 2);
  bf16* sWdT   = (bf16*)alloc((size_t)C_DIM * HS_DIM * 2);
  bf16* hsbuf  = (bf16*)alloc((size_t)S_TOK * HS_DIM * 2);
  bf16* hbuf   = (bf16*)alloc(((size_t)NSLOT + 16) * H_DIM * 2);
  float* yslot = (float*)alloc((size_t)NSLOT * C_DIM * 4);
  int*   topk_id    = (int*)alloc((size_t)NSLOT * 4);
  float* topk_w     = (float*)alloc((size_t)NSLOT * 4);
  int*   counts     = (int*)alloc(256);
  int*   offsets    = (int*)alloc(512);
  int*   cursors    = (int*)alloc(256);
  int*   sorted_tok = (int*)alloc((size_t)NSLOT * 4);
  float* slot_gate  = (float*)alloc((size_t)NSLOT * 4);
  int*   inv_slot   = (int*)alloc((size_t)NSLOT * 4);
  (void)ws_size; (void)in_sizes; (void)n_in; (void)out_size;

  hipMemsetAsync(counts, 0, E_NUM * sizeof(int), stream);

  // conversions (weights land in bf16, N-major)
  {
    int n = S_TOK * C_DIM;
    cvt_x_kernel<<<(n + 255) / 256, 256, 0, stream>>>(x, xb, n);
    size_t t1 = (size_t)E_NUM * C_DIM * H_DIM;
    cvt_transpose_kernel<<<(unsigned)((t1 + 255) / 256), 256, 0, stream>>>(Wg, WgT, E_NUM, C_DIM, H_DIM);
    cvt_transpose_kernel<<<(unsigned)((t1 + 255) / 256), 256, 0, stream>>>(Wu, WuT, E_NUM, C_DIM, H_DIM);
    cvt_transpose_kernel<<<(unsigned)((t1 + 255) / 256), 256, 0, stream>>>(Wd, WdT, E_NUM, H_DIM, C_DIM);
    size_t t2 = (size_t)C_DIM * HS_DIM;
    cvt_transpose_kernel<<<(unsigned)((t2 + 255) / 256), 256, 0, stream>>>(sWg, sWgT, 1, C_DIM, HS_DIM);
    cvt_transpose_kernel<<<(unsigned)((t2 + 255) / 256), 256, 0, stream>>>(sWu, sWuT, 1, C_DIM, HS_DIM);
    cvt_transpose_kernel<<<(unsigned)((t2 + 255) / 256), 256, 0, stream>>>(sWd, sWdT, 1, HS_DIM, C_DIM);
  }

  // router + dispatch
  router_kernel<<<S_TOK, E_NUM, 0, stream>>>(x, rw, bias, topk_id, topk_w, counts);
  scan_kernel<<<1, 32, 0, stream>>>(counts, offsets, cursors);
  scatter_kernel<<<(S_TOK + 255) / 256, 256, 0, stream>>>(topk_id, topk_w, cursors,
                                                          sorted_tok, slot_gate, inv_slot);

  // shared expert
  shared_up_kernel<<<dim3(S_TOK / 16, HS_DIM / 256), 256, 0, stream>>>(xb, sWgT, sWuT, hsbuf);
  shared_down_kernel<<<dim3(S_TOK / 16, C_DIM / 256), 256, 0, stream>>>(hsbuf, sWdT, out);

  // routed experts
  routed_up_kernel<<<dim3(E_NUM, H_DIM / 32), 256, 0, stream>>>(xb, WgT, WuT, offsets,
                                                                sorted_tok, slot_gate, hbuf);
  routed_down_kernel<<<dim3(E_NUM, C_DIM / 32), 256, 0, stream>>>(hbuf, WdT, offsets, yslot);

  // deterministic combine into out (out already holds the shared-expert term)
  combine_kernel<<<S_TOK, 128, 0, stream>>>(yslot, inv_slot, out);
}
